// KeyPointLineDetector_39161511805378
// MI455X (gfx1250) — matrix operand order repaired and checked by reading, not compiled, not run on hardware
//
#include <hip/hip_runtime.h>

// ---------------------------------------------------------------------------
// KeyPointLineDetector for MI455X (gfx1250, wave32, WMMA)
// Heavy GEMMs (convs as implicit GEMM + all linears) run on
// v_wmma_f32_16x16x32_f16 with f32 accumulation. Row-major-B GEMM tiles are
// staged into LDS by the Tensor Data Mover, double-buffered so the TDM
// transfer of tile t+1 overlaps the WMMA work on tile t.
// All GEMMs satisfy K % 32 == 0 (and BMODE-0 additionally N % 64 == 0).
// ---------------------------------------------------------------------------

typedef __attribute__((ext_vector_type(16))) _Float16 v16h;
typedef __attribute__((ext_vector_type(8)))  float    v8f;
typedef __attribute__((ext_vector_type(4)))  unsigned u32x4;
typedef __attribute__((ext_vector_type(8)))  int      i32x8;
typedef __attribute__((ext_vector_type(4)))  int      i32x4;

union FragH { v16h v; unsigned u[8]; _Float16 h[16]; };
union PackH4 { _Float16 h[4]; uint2 u2; };

// ===========================================================================
// WMMA GEMM: C[M,N] = act(A[M,K] * B[K,N] + bias)
//   BMODE 0: B row-major (ldb), tile staged via TENSOR_LOAD_TO_LDS (TDM).
//   BMODE 1: B = im2col(3x3, pad 1) of NCHW activations, spatial 128x128.
// Block: 128 threads (4 waves); each wave computes a 32x32 accumulator.
// Double-buffered LDS: stage tile t+1 while computing tile t.
// ===========================================================================
template <int BMODE>
__global__ __launch_bounds__(128) void wmma_gemm_kernel(
    const float* __restrict__ A, const float* __restrict__ Bm,
    const float* __restrict__ bias, float* __restrict__ C,
    int M, int N, int K, int lda, int ldb, int ldc,
    long long aBat, long long bBat, long long cBat,
    int act, int biasN)
{
  __shared__ _Float16 sA[2][64][40];   // A tiles, f16, padded rows (20 dwords)
  __shared__ _Float16 sB[2][64][40];   // B tiles transposed [n][k] (BMODE==1)
  __shared__ float    sBf[2][32][64];  // raw f32 B tiles from TDM (BMODE==0)

  const int tid  = threadIdx.x;
  const int bz   = blockIdx.z;
  const int row0 = blockIdx.y * 64;
  const int col0 = blockIdx.x * 64;
  const float* Ab = A  + (size_t)bz * aBat;
  const float* Bb = Bm + (size_t)bz * bBat;
  float*       Cb = C  + (size_t)bz * cBat;

  const int wave = tid >> 5, lane = tid & 31;
  const int wm = (wave >> 1) * 32;
  const int wn = (wave & 1) * 32;
  const int hv = lane >> 4, l15 = lane & 15;

  // ---- stage A tile (64x32) f32 -> f16: 4x b128 loads per thread ----
  // thread owns fixed 4-float column block k4 and rows r0, r0+16, +32, +48
  const int arow = tid >> 3;          // 0..15
  const int ak4  = (tid & 7) * 4;     // 0,4,...,28
  auto stageA = [&](int kb, int bs) {
    float4 tmp[4];
#pragma unroll
    for (int i = 0; i < 4; ++i) {
      int gm = row0 + arow + 16 * i;
      int gmc = gm < M ? gm : M - 1;       // clamped: load always legal
      tmp[i] = *(const float4*)(Ab + (size_t)gmc * lda + kb + ak4);
    }
#pragma unroll
    for (int i = 0; i < 4; ++i) {
      bool ok = (row0 + arow + 16 * i) < M;   // K%32==0: no k guard needed
      PackH4 pk;
      pk.h[0] = ok ? (_Float16)tmp[i].x : (_Float16)0.f;
      pk.h[1] = ok ? (_Float16)tmp[i].y : (_Float16)0.f;
      pk.h[2] = ok ? (_Float16)tmp[i].z : (_Float16)0.f;
      pk.h[3] = ok ? (_Float16)tmp[i].w : (_Float16)0.f;
      *(uint2*)&sA[bs][arow + 16 * i][ak4] = pk.u2;
    }
  };

  // ---- stage B tile ----
  // BMODE 1: thread owns fixed column n; k = kk0 + 2*i. h/w hoisted.
  const int bn   = tid & 63;
  const int bkk0 = tid >> 6;            // 0 or 1
  auto stageB = [&](int kb, int bs) {
    if constexpr (BMODE == 1) {
      const int gn = col0 + bn;
      const int h = gn >> 7, w = gn & 127;
      float tmp[16];
#pragma unroll
      for (int i = 0; i < 16; ++i) {
        int gk = kb + bkk0 + 2 * i;
        int ci = gk / 9, rem = gk - ci * 9;
        int kh = rem / 3, kw = rem - kh * 3;
        int y = h + kh - 1, x = w + kw - 1;
        int yc = y < 0 ? 0 : (y > 127 ? 127 : y);
        int xc = x < 0 ? 0 : (x > 127 ? 127 : x);
        tmp[i] = Bb[((size_t)ci * 128 + yc) * 128 + xc];
      }
#pragma unroll
      for (int i = 0; i < 16; ++i) {
        int k = bkk0 + 2 * i;
        int gk = kb + k;
        int ci = gk / 9, rem = gk - ci * 9;
        int kh = rem / 3, kw = rem - kh * 3;
        int y = h + kh - 1, x = w + kw - 1;
        bool ok = ((unsigned)y < 128u) & ((unsigned)x < 128u);
        sB[bs][bn][k] = ok ? (_Float16)tmp[i] : (_Float16)0.f;
      }
    } else {
      // TDM: wave 0 issues an async 2D tensor_load_to_lds of the 32x64 tile.
      if (tid < 32) {
        unsigned long long ga =
            (unsigned long long)(const void*)(Bb + (size_t)kb * ldb + col0);
        // flat shared address truncates to the LDS byte offset (ISA 10.2)
        unsigned ldsa = (unsigned)(unsigned long long)(void*)&sBf[bs][0][0];
        u32x4 g0;
        g0[0] = 1u;                                   // count=1, user mode
        g0[1] = ldsa;                                 // lds_addr
        g0[2] = (unsigned)(ga & 0xffffffffull);       // global_addr[31:0]
        g0[3] = (unsigned)((ga >> 32) & 0x01ffffffull) | (2u << 30); // type=2
        i32x8 g1;
        g1[0] = (int)(2u << 16);                      // data_size=4B, mask=0
        g1[1] = 0;                                    // tensor_dim0 lo16
        g1[2] = 0x4000;                               // tensor_dim0 hi16 (2^30)
        g1[3] = (int)(0x4000u | (64u << 16));         // dim1 hi16; tile_dim0=64
        g1[4] = 32;                                   // tile_dim1=32, dim2=0
        g1[5] = ldb;                                  // tensor_dim0_stride
        g1[6] = 0;
        g1[7] = 0;
        i32x4 g2; g2[0] = 0; g2[1] = 0; g2[2] = 0; g2[3] = 0;
        i32x4 g3; g3[0] = 0; g3[1] = 0; g3[2] = 0; g3[3] = 0;
#if defined(__clang_major__) && (__clang_major__ >= 23)
        i32x8 g4;
#pragma unroll
        for (int z = 0; z < 8; ++z) g4[z] = 0;
        __builtin_amdgcn_tensor_load_to_lds(g0, g1, g2, g3, g4, 0);
#else
        __builtin_amdgcn_tensor_load_to_lds(g0, g1, g2, g3, 0);
#endif
      }
    }
  };

  v8f acc[2][2];
#pragma unroll
  for (int i = 0; i < 2; ++i)
#pragma unroll
    for (int j = 0; j < 2; ++j)
#pragma unroll
      for (int e = 0; e < 8; ++e) acc[i][j][e] = 0.f;

  const int T = (K + 31) >> 5;
  stageA(0, 0);
  stageB(0, 0);

  for (int t = 0; t < T; ++t) {
    const int bs = t & 1;
    if constexpr (BMODE == 0) {
      // the TDM op filling buffer bs is the oldest outstanding one
      if (tid < 32) __builtin_amdgcn_s_wait_tensorcnt(0);
    }
    __syncthreads();   // tile t fully staged & visible to all waves
    if (t + 1 < T) {   // overlap: stage tile t+1 while computing tile t
      stageA((t + 1) << 5, bs ^ 1);
      stageB((t + 1) << 5, bs ^ 1);
      if (t + 2 < T) {  // warm GL2 for tile t+2 A rows (global_prefetch_b8)
        int gm = row0 + arow;
        if (gm < M)
          __builtin_prefetch(&Ab[(size_t)gm * lda + ((t + 2) << 5) + ak4], 0, 1);
      }
    }

    // ---- load fragments (k = 16*(lane/16) + s, same order in A and B) and multiply ----
    FragH af[2], bf[2];
    const unsigned* sAu = (const unsigned*)&sA[bs][0][0];
#pragma unroll
    for (int i = 0; i < 2; ++i) {
      int m = wm + i * 16 + l15;
#pragma unroll
      for (int vv = 0; vv < 8; ++vv) {
        int k = hv * 16 + 2 * vv;
        af[i].u[vv] = sAu[m * 20 + (k >> 1)];
      }
    }
    if constexpr (BMODE == 1) {
      const unsigned* sBu = (const unsigned*)&sB[bs][0][0];
#pragma unroll
      for (int j = 0; j < 2; ++j) {
        int n = wn + j * 16 + l15;
#pragma unroll
        for (int vv = 0; vv < 8; ++vv) {
          int k = hv * 16 + 2 * vv;
          bf[j].u[vv] = sBu[n * 20 + (k >> 1)];
        }
      }
    } else {
#pragma unroll
      for (int j = 0; j < 2; ++j) {
        int n = wn + j * 16 + l15;
#pragma unroll
        for (int vv = 0; vv < 8; ++vv) {
          int k = hv * 16 + 2 * vv;
          bf[j].h[2 * vv + 0] = (_Float16)sBf[bs][k + 0][n];
          bf[j].h[2 * vv + 1] = (_Float16)sBf[bs][k + 1][n];
        }
      }
    }
#pragma unroll
    for (int i = 0; i < 2; ++i)
#pragma unroll
      for (int j = 0; j < 2; ++j)
        acc[i][j] = __builtin_amdgcn_wmma_f32_16x16x32_f16(
            false, af[i].v, false, bf[j].v, (short)0, acc[i][j], false, false);
  }

  // ---- epilogue: bias + activation + store ----
#pragma unroll
  for (int i = 0; i < 2; ++i) {
#pragma unroll
    for (int j = 0; j < 2; ++j) {
      int mB = row0 + wm + i * 16 + hv * 8;
      int nn = col0 + wn + j * 16 + l15;
#pragma unroll
      for (int rr = 0; rr < 8; ++rr) {
        int m = mB + rr;
        if (m < M && nn < N) {
          float v = acc[i][j][rr];
          if (bias) v += biasN ? bias[nn] : bias[m];
          if (act == 1)      v = fmaxf(v, 0.f);
          else if (act == 2) v = 1.f / (1.f + __expf(-v));
          else if (act == 3) v = tanhf(v);
          Cb[(size_t)m * ldc + nn] = v;
        }
      }
    }
  }
}

// ===========================================================================
// Elementwise / glue kernels
// ===========================================================================
__global__ void upsample_add_kernel(float* __restrict__ dst,
                                    const float* __restrict__ src,
                                    int C, int H, int W, int total)
{
  int idx = blockIdx.x * 256 + threadIdx.x;
  if (idx >= total) return;
  int w = idx % W;
  int h = (idx / W) % H;
  int c = (idx / (W * H)) % C;
  int b = idx / (W * H * C);
  int Hs = H >> 1, Ws = W >> 1;
  dst[idx] += src[(((size_t)b * C + c) * Hs + (h >> 1)) * Ws + (w >> 1)];
}

__global__ void nms_kernel(const float* __restrict__ hm, float* __restrict__ out,
                           int total)
{
  int idx = blockIdx.x * 256 + threadIdx.x;
  if (idx >= total) return;
  int w = idx & 127;
  int h = (idx >> 7) & 127;
  const float* plane = hm + (idx & ~16383);
  float mx = -1e30f;
#pragma unroll
  for (int dy = -1; dy <= 1; ++dy)
#pragma unroll
    for (int dx = -1; dx <= 1; ++dx) {
      int y = h + dy, x = w + dx;
      if ((unsigned)y < 128u && (unsigned)x < 128u)
        mx = fmaxf(mx, plane[y * 128 + x]);
    }
  float v = hm[idx];
  out[idx] = (v == mx) ? v : 0.f;
}

// one block per (class, batch): iterative top-12 + ref-point build
__global__ void topk_ref_kernel(const float* __restrict__ nms,
                                const float* __restrict__ off2,
                                float* __restrict__ ref)
{
  int c = blockIdx.x, b = blockIdx.y, t = threadIdx.x;
  const float* src = nms + ((size_t)b * 32 + c) * 16384;
  __shared__ float sval[256];
  __shared__ int   sidx[256];
  __shared__ int   sel[12];
  for (int kk = 0; kk < 12; ++kk) {
    float best = -1e30f; int bi = 1 << 30;
    for (int i = t; i < 16384; i += 256) {
      bool taken = false;
      for (int s = 0; s < kk; ++s) taken |= (sel[s] == i);
      float v = taken ? -1e30f : src[i];
      if (v > best || (v == best && i < bi)) { best = v; bi = i; }
    }
    sval[t] = best; sidx[t] = bi;
    __syncthreads();
    for (int s = 128; s > 0; s >>= 1) {
      if (t < s) {
        if (sval[t + s] > sval[t] ||
            (sval[t + s] == sval[t] && sidx[t + s] < sidx[t])) {
          sval[t] = sval[t + s]; sidx[t] = sidx[t + s];
        }
      }
      __syncthreads();
    }
    if (t == 0) {
      int ind = sidx[0];
      sel[kk] = ind;
      float ys = (float)(ind >> 7), xs = (float)(ind & 127);
      float ox = off2[((size_t)b * 64 + c * 2 + 0) * 16384 + ind];
      float oy = off2[((size_t)b * 64 + c * 2 + 1) * 16384 + ind];
      float px = fminf(fmaxf((xs + ox) * (1.f / 128.f), 0.f), 1.f);
      float py = fminf(fmaxf((ys + oy) * (1.f / 128.f), 0.f), 1.f);
      int n = c * 12 + kk;
      ref[((size_t)b * 384 + n) * 2 + 0] = px;
      ref[((size_t)b * 384 + n) * 2 + 1] = py;
    }
    __syncthreads();
  }
}

__global__ void build_memory_kernel(const float* __restrict__ feat,
                                    const float* __restrict__ colE,
                                    const float* __restrict__ rowE,
                                    float* __restrict__ mem, int total)
{
  int idx = blockIdx.x * 256 + threadIdx.x;
  if (idx >= total) return;
  int d = idx & 255;
  int hw = (idx >> 8) & 16383;
  int b = idx >> 22;
  int h = hw >> 7, w = hw & 127;
  float pos = (d < 128) ? colE[w * 128 + d] : rowE[h * 128 + (d - 128)];
  mem[idx] = feat[(((size_t)b * 256 + d) * 128 + h) * 128 + w] + pos;
}

__global__ void query_init_kernel(const float* __restrict__ qe,
                                  float* __restrict__ q, int total)
{
  int idx = blockIdx.x * 256 + threadIdx.x;
  if (idx >= total) return;
  int d = idx & 255;
  int n = (idx >> 8) % 384;
  q[idx] = qe[n * 256 + d];
}

// online-softmax MHA over 384 queries; one thread per (b, n, head)
__global__ void self_attn_kernel(const float* __restrict__ qkv,
                                 float* __restrict__ out)
{
  int idx = blockIdx.x * 256 + threadIdx.x;
  if (idx >= 4 * 384 * 8) return;
  int hd = idx & 7;
  int n = (idx >> 3) % 384;
  int b = idx / (384 * 8);
  const float* qrow = qkv + ((size_t)(b * 384 + n) * 768) + hd * 32;
  float q[32], accv[32];
#pragma unroll
  for (int d = 0; d < 32; ++d) { q[d] = qrow[d] * 0.17677669529663687f; accv[d] = 0.f; }
  float m = -1e30f, l = 0.f;
  for (int j = 0; j < 384; ++j) {
    const float* krow = qkv + ((size_t)(b * 384 + j) * 768) + 256 + hd * 32;
    float s = 0.f;
#pragma unroll
    for (int d = 0; d < 32; ++d) s += q[d] * krow[d];
    float mn = fmaxf(m, s);
    float corr = __expf(m - mn), p = __expf(s - mn);
    l = l * corr + p;
    const float* vrow = qkv + ((size_t)(b * 384 + j) * 768) + 512 + hd * 32;
#pragma unroll
    for (int d = 0; d < 32; ++d) accv[d] = accv[d] * corr + p * vrow[d];
    m = mn;
  }
  float inv = 1.f / l;
  float* orow = out + (size_t)(b * 384 + n) * 256 + hd * 32;
#pragma unroll
  for (int d = 0; d < 32; ++d) orow[d] = accv[d] * inv;
}

// deformable sampling; one thread per (b, n, head, dim)
__global__ void deform_kernel(const float* __restrict__ val,
                              const float* __restrict__ ref,
                              const float* __restrict__ offq,
                              const float* __restrict__ awq,
                              float* __restrict__ out)
{
  int idx = blockIdx.x * 256 + threadIdx.x;
  if (idx >= 4 * 384 * 8 * 32) return;
  int d = idx & 31;
  int hd = (idx >> 5) & 7;
  int r = idx >> 8;            // b*384 + n
  int b = r / 384;
  float rx = ref[r * 2 + 0], ry = ref[r * 2 + 1];
  float e[4];
  float mx = -1e30f;
#pragma unroll
  for (int p = 0; p < 4; ++p) { e[p] = awq[r * 32 + hd * 4 + p]; mx = fmaxf(mx, e[p]); }
  float den = 0.f;
#pragma unroll
  for (int p = 0; p < 4; ++p) { e[p] = __expf(e[p] - mx); den += e[p]; }
  float inv = 1.f / den;
  const float* vb = val + (size_t)b * 16384 * 256 + hd * 32 + d;
  float accv = 0.f;
#pragma unroll
  for (int p = 0; p < 4; ++p) {
    float ox = offq[r * 64 + hd * 8 + p * 2 + 0] * 0.5f;
    float oy = offq[r * 64 + hd * 8 + p * 2 + 1] * 0.5f;
    float x = (rx + ox) * 128.f - 0.5f;
    float y = (ry + oy) * 128.f - 0.5f;
    float x0f = floorf(x), y0f = floorf(y);
    int x0 = (int)x0f, y0 = (int)y0f;
    float fx = x - x0f, fy = y - y0f;
    float wp = e[p] * inv;
#pragma unroll
    for (int dy = 0; dy < 2; ++dy)
#pragma unroll
      for (int dx = 0; dx < 2; ++dx) {
        int xi = x0 + dx, yi = y0 + dy;
        if ((unsigned)xi < 128u && (unsigned)yi < 128u) {
          float wgt = (dx ? fx : 1.f - fx) * (dy ? fy : 1.f - fy);
          accv += wp * wgt * vb[(size_t)(yi * 128 + xi) * 256];
        }
      }
  }
  out[(size_t)r * 256 + hd * 32 + d] = accv;
}

// out = LN(x + r) * g + b ; one 256-thread block per row (D=256)
__global__ void ln_residual_kernel(const float* __restrict__ x,
                                   const float* __restrict__ r,
                                   const float* __restrict__ g,
                                   const float* __restrict__ bb,
                                   float* __restrict__ out)
{
  int row = blockIdx.x, t = threadIdx.x;
  __shared__ float red[256];
  float v = x[(size_t)row * 256 + t] + r[(size_t)row * 256 + t];
  red[t] = v;
  __syncthreads();
  for (int s = 128; s > 0; s >>= 1) { if (t < s) red[t] += red[t + s]; __syncthreads(); }
  float mean = red[0] * (1.f / 256.f);
  __syncthreads();
  float c = v - mean;
  red[t] = c * c;
  __syncthreads();
  for (int s = 128; s > 0; s >>= 1) { if (t < s) red[t] += red[t + s]; __syncthreads(); }
  float var = red[0] * (1.f / 256.f);
  out[(size_t)row * 256 + t] = c * rsqrtf(var + 1e-5f) * g[t] + bb[t];
}

// out[r,o] = act(sum_k x[r,k] * w[k,o] + b[o]) for small dout
__global__ void small_linear_kernel(const float* __restrict__ x,
                                    const float* __restrict__ w,
                                    const float* __restrict__ b,
                                    float* __restrict__ out,
                                    int rows, int din, int dout, int act)
{
  int idx = blockIdx.x * 256 + threadIdx.x;
  if (idx >= rows * dout) return;
  int r = idx / dout, o = idx - r * dout;
  float s = b ? b[o] : 0.f;
  const float* xr = x + (size_t)r * din;
  for (int k = 0; k < din; ++k) s += xr[k] * w[(size_t)k * dout + o];
  if (act == 1)      s = fmaxf(s, 0.f);
  else if (act == 2) s = 1.f / (1.f + __expf(-s));
  out[idx] = s;
}

__global__ void ref_update_kernel(float* __restrict__ ref,
                                  const float* __restrict__ rh, int total)
{
  int idx = blockIdx.x * 256 + threadIdx.x;
  if (idx >= total) return;
  float v = ref[idx] + (rh[idx] - 0.5f) * 0.1f;
  ref[idx] = fminf(fmaxf(v, 0.f), 1.f);
}

__global__ void line_post_kernel(const float* __restrict__ lp,
                                 float* __restrict__ out, int total)
{
  int idx = blockIdx.x * 256 + threadIdx.x;
  if (idx >= total) return;
  int c = idx & 3;
  float v = lp[idx];
  out[idx] = (c == 0) ? tanhf(v) * 3.14159265358979f
                      : 1.f / (1.f + __expf(-v));
}

// ===========================================================================
// Host orchestration
// ===========================================================================
static inline int nb256(long long n) { return (int)((n + 255) / 256); }

static void launch_gemm(hipStream_t st, const float* A, const float* B,
                        const float* bias, float* C, int M, int N, int K,
                        int lda, int ldb, int ldc, long long aB, long long bB,
                        long long cB, int batch, int bmode, int act, int biasN)
{
  dim3 grid((N + 63) / 64, (M + 63) / 64, batch), blk(128);
  if (bmode == 0)
    wmma_gemm_kernel<0><<<grid, blk, 0, st>>>(A, B, bias, C, M, N, K, lda, ldb,
                                              ldc, aB, bB, cB, act, biasN);
  else
    wmma_gemm_kernel<1><<<grid, blk, 0, st>>>(A, B, bias, C, M, N, K, lda, ldb,
                                              ldc, aB, bB, cB, act, biasN);
}

extern "C" void kernel_launch(void* const* d_in, const int* in_sizes, int n_in,
                              void* d_out, int out_size, void* d_ws, size_t ws_size,
                              hipStream_t stream)
{
  (void)in_sizes; (void)n_in; (void)out_size; (void)ws_size;
  auto F = [&](int i) { return (const float*)d_in[i]; };

  const float* feats[4] = { F(0), F(1), F(2), F(3) };
  // --- params (JAX pytree flatten: dict keys sorted; {'w','b'} -> b, w) ---
  const float* col_embed = F(4);
  const float* conf1_b = F(5);  const float* conf1_w = F(6);
  const float* conf2_b = F(7);  const float* conf2_w = F(8);
  const float* fpn_b0 = F(9);   const float* fpn_w0 = F(10);
  const float* hm1_b = F(17);   const float* hm1_w = F(18);
  const float* hm2_b = F(19);   const float* hm2_w = F(20);
  const float* kp1_b = F(21);   const float* kp1_w = F(22);
  const float* kp2_b = F(23);   const float* kp2_w = F(24);
  const float* lat_b[4]; const float* lat_w[4];
  for (int i = 0; i < 4; ++i) { lat_b[i] = F(25 + 2 * i); lat_w[i] = F(26 + 2 * i); }
  const float* line1_b = F(177); const float* line1_w = F(178);
  const float* line2_b = F(179); const float* line2_w = F(180);
  const float* off1_b = F(181);  const float* off1_w = F(182);
  const float* off2_b = F(183);  const float* off2_w = F(184);
  const float* query_embed = F(185);
  const float* row_embed   = F(186);
  const float* vis1_b = F(187); const float* vis1_w = F(188);
  const float* vis2_b = F(189); const float* vis2_w = F(190);

  // --- workspace bump allocator ---
  float* ws = (float*)d_ws;
  size_t off = 0;
  auto alloc = [&](size_t n) { float* p = ws + off; off += (n + 63) & ~(size_t)63; return p; };
  float* lat0    = alloc(4ull * 256 * 16384);
  float* lat1    = alloc(4ull * 256 * 4096);
  float* lat2    = alloc(4ull * 256 * 1024);
  float* lat3    = alloc(4ull * 256 * 256);
  float* feat    = alloc(4ull * 256 * 16384);
  float* off2buf = alloc(4ull * 64 * 16384);
  float* nmsbuf  = alloc(4ull * 32 * 16384);
  float* refbuf  = alloc(4ull * 384 * 2);
  float* memory  = alloc(4ull * 16384 * 256);
  float* query   = alloc(1536ull * 256);
  float* qkvbuf  = alloc(1536ull * 768);
  float* obuf    = alloc(1536ull * 256);
  float* dbuf    = alloc(1536ull * 256);
  float* offq    = alloc(1536ull * 64);
  float* awq     = alloc(1536ull * 32);
  float* ffbuf   = alloc(1536ull * 1024);
  float* rhbuf   = alloc(1536ull * 2);
  float* l1buf   = alloc(128ull * 256);
  float* lpbuf   = alloc(128ull * 4);
  float* kpbuf   = alloc(1536ull * 256);
  float* visbuf  = alloc(1536ull * 128);
  float* confbuf = alloc(128ull * 128);
  float* hbuf    = lat0;   // reused after fpn conv consumes lat0
  float* valbuf  = feat;   // reused after memory is built

  // --- output regions (tuple order: heatmap, kps, line_params, vis, conf) ---
  float* out_hm   = (float*)d_out;
  float* out_kps  = out_hm  + 4ull * 32 * 16384;
  float* out_lp   = out_kps + 4ull * 32 * 12 * 2;
  float* out_vis  = out_lp  + 4ull * 32 * 4;
  float* out_conf = out_vis + 4ull * 32 * 12;

  const int CHS[4] = { 256, 512, 1024, 2048 };
  const int HWs[4] = { 16384, 4096, 1024, 256 };
  const int Hs[4]  = { 128, 64, 32, 16 };
  float* lats[4] = { lat0, lat1, lat2, lat3 };

  // ---- FPN laterals (1x1 conv = GEMM, TDM-staged B) ----
  for (int i = 0; i < 4; ++i)
    launch_gemm(stream, lat_w[i], feats[i], lat_b[i], lats[i],
                256, HWs[i], CHS[i], CHS[i], HWs[i], HWs[i],
                0, (long long)CHS[i] * HWs[i], 256ll * HWs[i], 4, 0, 0, 0);
  // ---- top-down upsample-add ----
  for (int i = 3; i > 0; --i) {
    int total = 4 * 256 * HWs[i - 1];
    upsample_add_kernel<<<nb256(total), 256, 0, stream>>>(
        lats[i - 1], lats[i], 256, Hs[i - 1], Hs[i - 1], total);
  }
  // ---- fpn[0] 3x3 conv (implicit GEMM, K=2304) ----
  launch_gemm(stream, fpn_w0, lat0, fpn_b0, feat, 256, 16384, 2304,
              2304, 0, 16384, 0, 256ll * 16384, 256ll * 16384, 4, 1, 0, 0);
  // ---- heatmap head ----
  launch_gemm(stream, hm1_w, feat, hm1_b, hbuf, 256, 16384, 2304,
              2304, 0, 16384, 0, 256ll * 16384, 256ll * 16384, 4, 1, 1, 0);
  launch_gemm(stream, hm2_w, hbuf, hm2_b, out_hm, 32, 16384, 256,
              256, 16384, 16384, 0, 256ll * 16384, 32ll * 16384, 4, 0, 2, 0);
  // ---- offset head ----
  launch_gemm(stream, off1_w, feat, off1_b, hbuf, 256, 16384, 2304,
              2304, 0, 16384, 0, 256ll * 16384, 256ll * 16384, 4, 1, 1, 0);
  launch_gemm(stream, off2_w, hbuf, off2_b, off2buf, 64, 16384, 256,
              256, 16384, 16384, 0, 256ll * 16384, 64ll * 16384, 4, 0, 0, 0);
  // ---- NMS + top-K + reference points ----
  nms_kernel<<<nb256(4 * 32 * 16384), 256, 0, stream>>>(out_hm, nmsbuf, 4 * 32 * 16384);
  topk_ref_kernel<<<dim3(32, 4), 256, 0, stream>>>(nmsbuf, off2buf, refbuf);
  // ---- memory (feat + positional embedding) and query init ----
  build_memory_kernel<<<nb256(4ll * 16384 * 256), 256, 0, stream>>>(
      feat, col_embed, row_embed, memory, 4 * 16384 * 256);
  query_init_kernel<<<nb256(1536 * 256), 256, 0, stream>>>(query_embed, query, 1536 * 256);

  // ---- decoder ----
  for (int L = 0; L < 6; ++L) {
    int base = 33 + L * 24;
    const float *ao_b = F(base + 0),  *ao_w = F(base + 1);
    const float *aw_b = F(base + 2),  *aw_w = F(base + 3);
    const float *f1_b = F(base + 4),  *f1_w = F(base + 5);
    const float *f2_b = F(base + 6),  *f2_w = F(base + 7);
    const float *n1_b = F(base + 8),  *n1_g = F(base + 9);
    const float *n2_b = F(base + 10), *n2_g = F(base + 11);
    const float *n3_b = F(base + 12), *n3_g = F(base + 13);
    const float *op_b = F(base + 14), *op_w = F(base + 15);
    const float *qk_b = F(base + 16), *qk_w = F(base + 17);
    const float *rh_b = F(base + 18), *rh_w = F(base + 19);
    const float *so_b = F(base + 20), *so_w = F(base + 21);
    const float *vp_b = F(base + 22), *vp_w = F(base + 23);

    // self-attention
    launch_gemm(stream, query, qk_w, qk_b, qkvbuf, 1536, 768, 256,
                256, 768, 768, 0, 0, 0, 1, 0, 0, 1);
    self_attn_kernel<<<nb256(4 * 384 * 8), 256, 0, stream>>>(qkvbuf, obuf);
    launch_gemm(stream, obuf, ao_w, ao_b, dbuf, 1536, 256, 256,
                256, 256, 256, 0, 0, 0, 1, 0, 0, 1);
    ln_residual_kernel<<<1536, 256, 0, stream>>>(query, dbuf, n1_g, n1_b, query);

    // deformable attention
    launch_gemm(stream, memory, vp_w, vp_b, valbuf, 65536, 256, 256,
                256, 256, 256, 0, 0, 0, 1, 0, 0, 1);
    launch_gemm(stream, query, so_w, so_b, offq, 1536, 64, 256,
                256, 64, 64, 0, 0, 0, 1, 0, 3, 1);                 // tanh fused
    small_linear_kernel<<<nb256(1536 * 32), 256, 0, stream>>>(
        query, aw_w, aw_b, awq, 1536, 256, 32, 0);
    deform_kernel<<<nb256(4 * 384 * 8 * 32), 256, 0, stream>>>(
        valbuf, refbuf, offq, awq, obuf);
    launch_gemm(stream, obuf, op_w, op_b, dbuf, 1536, 256, 256,
                256, 256, 256, 0, 0, 0, 1, 0, 0, 1);
    ln_residual_kernel<<<1536, 256, 0, stream>>>(query, dbuf, n2_g, n2_b, query);

    // FFN
    launch_gemm(stream, query, f1_w, f1_b, ffbuf, 1536, 1024, 256,
                256, 1024, 1024, 0, 0, 0, 1, 0, 1, 1);             // relu fused
    launch_gemm(stream, ffbuf, f2_w, f2_b, dbuf, 1536, 256, 1024,
                1024, 256, 256, 0, 0, 0, 1, 0, 0, 1);
    ln_residual_kernel<<<1536, 256, 0, stream>>>(query, dbuf, n3_g, n3_b, query);

    // reference refinement
    small_linear_kernel<<<nb256(1536 * 2), 256, 0, stream>>>(
        query, rh_w, rh_b, rhbuf, 1536, 256, 2, 2);                // sigmoid
    ref_update_kernel<<<nb256(4 * 384 * 2), 256, 0, stream>>>(refbuf, rhbuf, 4 * 384 * 2);
  }

  // ---- heads ----
  launch_gemm(stream, query, line1_w, line1_b, l1buf, 128, 256, 3072,
              3072, 256, 256, 0, 0, 0, 1, 0, 1, 1);
  small_linear_kernel<<<nb256(128 * 4), 256, 0, stream>>>(
      l1buf, line2_w, line2_b, lpbuf, 128, 256, 4, 0);
  line_post_kernel<<<nb256(512), 256, 0, stream>>>(lpbuf, out_lp, 512);

  launch_gemm(stream, query, kp1_w, kp1_b, kpbuf, 1536, 256, 256,
              256, 256, 256, 0, 0, 0, 1, 0, 1, 1);
  small_linear_kernel<<<nb256(1536 * 2), 256, 0, stream>>>(
      kpbuf, kp2_w, kp2_b, out_kps, 1536, 256, 2, 2);

  launch_gemm(stream, query, vis1_w, vis1_b, visbuf, 1536, 128, 256,
              256, 128, 128, 0, 0, 0, 1, 0, 1, 1);
  small_linear_kernel<<<nb256(1536), 256, 0, stream>>>(
      visbuf, vis2_w, vis2_b, out_vis, 1536, 128, 1, 0);

  launch_gemm(stream, query, conf1_w, conf1_b, confbuf, 128, 128, 3072,
              3072, 128, 128, 0, 0, 0, 1, 0, 1, 1);
  small_linear_kernel<<<nb256(128), 256, 0, stream>>>(
      confbuf, conf2_w, conf2_b, out_conf, 128, 128, 1, 0);
}
